// ReferenceCFConv_38328288150131
// MI455X (gfx1250) — compile-verified, matrix operand order checked by reading, not run on hardware
//
#include <hip/hip_runtime.h>

// CFConv (SchNet-style continuous-filter conv + edge softmax attention) for gfx1250.
// Edge MLP runs on v_wmma_f32_16x16x32_f16: inputs are row-normalized (|v|<=1) or
// tanh outputs (|v|<=1), so f16 A/B with f32 accumulate is numerically sound.

#define N_NODES 32768
#define N_EDGES 524288
#define IN_CH   128
#define OUT_CH  128
#define EDGE_CH 64
#define HIDDEN  128
#define NGROUPS (N_EDGES / 16)

typedef __attribute__((ext_vector_type(16))) _Float16 v16h;
typedef __attribute__((ext_vector_type(8)))  float    v8f;

// Branch-free tanh: native v_tanh_f32 on gfx1250 (probe-confirmed in round 2 disasm).
__device__ __forceinline__ float fast_tanh(float v) {
#if __has_builtin(__builtin_amdgcn_tanhf)
  return __builtin_amdgcn_tanhf(v);
#elif __has_builtin(__builtin_amdgcn_tanh_f32)
  return __builtin_amdgcn_tanh_f32(v);
#else
  const float ex = __expf(2.0f * v);
  return (ex - 1.0f) * __builtin_amdgcn_rcpf(ex + 1.0f);
#endif
}

// Monotone float atomic-max via integer atomics (works with -inf init).
__device__ __forceinline__ void atomicMaxF(float* a, float v) {
  if (v >= 0.0f) atomicMax((int*)a, __float_as_int(v));
  else           atomicMin((unsigned int*)a, __float_as_uint(v));
}

__global__ __launch_bounds__(256) void cfconv_init(float* __restrict__ out,
                                                   float* __restrict__ seg_max,
                                                   float* __restrict__ denom) {
  const int i = blockIdx.x * blockDim.x + threadIdx.x;
  if (i < N_NODES * OUT_CH) out[i] = 0.0f;
  if (i < N_NODES) { seg_max[i] = -__builtin_inff(); denom[i] = 0.0f; }
}

__global__ __launch_bounds__(256) void cfconv_mid(const int* __restrict__ eidx,
                                                  float* __restrict__ ws_score,
                                                  const float* __restrict__ seg_max,
                                                  float* __restrict__ denom) {
  const int e = blockIdx.x * blockDim.x + threadIdx.x;
  if (e >= N_EDGES) return;
  const int t = eidx[N_EDGES + e];        // target row of edge_index
  float m = seg_max[t];
  if (m == -__builtin_inff()) m = 0.0f;   // reference: where(isfinite, max, 0)
  const float ev = __expf(ws_score[e] - m);
  ws_score[e] = ev;                       // overwrite score with exp-value
  atomicAdd(&denom[t], ev);
}

// PHASE 0: scores + segment max.  PHASE 1: weighted scatter-add + attn weights out.
template <int PHASE>
__global__ __launch_bounds__(256) void cfconv_heavy(
    const float* __restrict__ x,  const int* __restrict__ eidx,
    const float* __restrict__ ea, const float* __restrict__ W1,
    const float* __restrict__ b1, const float* __restrict__ W2,
    const float* __restrict__ b2, const float* __restrict__ attn,
    float* __restrict__ ws_score, float* __restrict__ seg_max,
    const float* __restrict__ denom, float* __restrict__ out) {
  __shared__ __align__(32) _Float16 a_norm[16][EDGE_CH];  // normalized edge attrs (f16)
  __shared__ __align__(32) _Float16 h_tile[16][HIDDEN];   // tanh(layer1) (f16)
  __shared__ float s_scores[16];
  __shared__ float s_w[16];
  __shared__ int   s_src[16], s_tgt[16];

  const int tid   = threadIdx.x;
  const int lane  = tid & 31;
  const int wid   = tid >> 5;          // n-tile id 0..7 (8 waves per block)
  const int colL  = lane & 15;
  const int hi    = lane >> 4;         // 0 = lanes 0-15, 1 = lanes 16-31
  const int col   = wid * 16 + colL;   // output channel owned by this lane
  const int krow0 = hi * 16;

  // ---- Resident B-matrix fragments (loaded once, reused for all edge groups) ----
  // B (32x16 f16, this code's K order): lane%16 = column, lanes 0-15 hold K=0..15, 16-31 hold K=16..31.
  v16h bW1[2], bW2[4];
#pragma unroll
  for (int kb = 0; kb < 2; ++kb)
#pragma unroll
    for (int i = 0; i < 16; ++i)
      bW1[kb][i] = (_Float16)W1[(kb * 32 + krow0 + i) * HIDDEN + col];
#pragma unroll
  for (int kb = 0; kb < 4; ++kb)
#pragma unroll
    for (int i = 0; i < 16; ++i)
      bW2[kb][i] = (_Float16)W2[(kb * 32 + krow0 + i) * OUT_CH + col];
  const float b1v = b1[col], b2v = b2[col], attnv = attn[col];

  for (int g = blockIdx.x; g < NGROUPS; g += gridDim.x) {
    const int e0 = g * 16;
    __syncthreads();  // previous iteration fully consumed LDS

    // ---- Stage: row-normalize 16 edges x 64ch into f16 LDS; fetch indices ----
    {
      const int e  = tid >> 4;           // 16 threads per edge
      const int cg = (tid & 15) * 4;
      const float4 av = *(const float4*)&ea[(long)(e0 + e) * EDGE_CH + cg];
      float ss = av.x * av.x + av.y * av.y + av.z * av.z + av.w * av.w;
#pragma unroll
      for (int off = 8; off > 0; off >>= 1) ss += __shfl_xor(ss, off, 16);
      const float inv = 1.0f / (sqrtf(ss) + 1e-8f);
      a_norm[e][cg + 0] = (_Float16)(av.x * inv);
      a_norm[e][cg + 1] = (_Float16)(av.y * inv);
      a_norm[e][cg + 2] = (_Float16)(av.z * inv);
      a_norm[e][cg + 3] = (_Float16)(av.w * inv);
      if (tid < 16) {
        s_src[tid] = eidx[e0 + tid];
        const int t = eidx[N_EDGES + e0 + tid];
        s_tgt[tid] = t;
        if (PHASE == 0) {
          s_scores[tid] = 0.0f;
        } else {
          const float ev = ws_score[e0 + tid];                 // exp-values from mid pass
          const float w  = ev / (denom[t] + 1e-16f);
          s_w[tid] = w;
          out[(unsigned)N_NODES * OUT_CH + e0 + tid] = w;      // attention_weights output
        }
      }
    }
    __syncthreads();

    // ---- Layer 1: h = tanh(a_norm @ W1 + b1), K=64 -> 2 WMMAs ----
    v8f hacc;
#pragma unroll
    for (int r = 0; r < 8; ++r) hacc[r] = b1v;
    {
      // A (16x32 f16, this code's K order, same as B): lane = row (colL), halves = K {krow0..krow0+15}
      v16h a0 = *(const v16h*)&a_norm[colL][krow0];
      v16h a1 = *(const v16h*)&a_norm[colL][32 + krow0];
      hacc = __builtin_amdgcn_wmma_f32_16x16x32_f16(false, a0, false, bW1[0],
                                                    (short)0, hacc, false, false);
      hacc = __builtin_amdgcn_wmma_f32_16x16x32_f16(false, a1, false, bW1[1],
                                                    (short)0, hacc, false, false);
    }
#pragma unroll
    for (int r = 0; r < 8; ++r) {            // C/D layout: vgpr r -> row r + 8*hi
      h_tile[r + hi * 8][col] = (_Float16)fast_tanh(hacc[r]);
    }
    __syncthreads();

    // ---- Layer 2: filters = h @ W2 + b2, K=128 -> 4 WMMAs ----
    v8f facc;
#pragma unroll
    for (int r = 0; r < 8; ++r) facc[r] = b2v;
#pragma unroll
    for (int kb = 0; kb < 4; ++kb) {
      v16h a2 = *(const v16h*)&h_tile[colL][kb * 32 + krow0];
      facc = __builtin_amdgcn_wmma_f32_16x16x32_f16(false, a2, false, bW2[kb],
                                                    (short)0, facc, false, false);
    }

    // ---- messages = x[src] * filters (x is 16MB -> L2-resident gather) ----
    // 32-bit unsigned element offsets (< 2^23) -> GVS addressing with scale_offset,
    // avoiding per-row 64-bit address math.
    float msg[8];
#pragma unroll
    for (int r = 0; r < 8; ++r) {
      const int m = r + hi * 8;
      const unsigned xoff = (unsigned)s_src[m] * IN_CH + (unsigned)col;
      msg[r] = x[xoff] * facc[r];
    }

    if (PHASE == 0) {
      // scores[e] = messages[e] . attn : reduce over 16 lanes (channels of this tile)
#pragma unroll
      for (int r = 0; r < 8; ++r) {
        float p = msg[r] * attnv;
#pragma unroll
        for (int off = 8; off > 0; off >>= 1) p += __shfl_xor(p, off, 16);
        if (colL == 0) atomicAdd(&s_scores[r + hi * 8], p);  // 8 n-tiles accumulate
      }
      __syncthreads();
      if (tid < 16) {
        const float sc = s_scores[tid];
        ws_score[e0 + tid] = sc;
        atomicMaxF(&seg_max[s_tgt[tid]], sc);
      }
    } else {
      // weighted scatter-add into node outputs (16MB region -> L2 atomics)
#pragma unroll
      for (int r = 0; r < 8; ++r) {
        const int m = r + hi * 8;
        const unsigned ooff = (unsigned)s_tgt[m] * OUT_CH + (unsigned)col;
        atomicAdd(&out[ooff], msg[r] * s_w[m]);
      }
    }
  }
}

extern "C" void kernel_launch(void* const* d_in, const int* in_sizes, int n_in,
                              void* d_out, int out_size, void* d_ws, size_t ws_size,
                              hipStream_t stream) {
  const float* x    = (const float*)d_in[0];
  const int*   eidx = (const int*)d_in[1];   // edge_index [2, E]
  const float* ea   = (const float*)d_in[2];
  const float* W1   = (const float*)d_in[3];
  const float* b1   = (const float*)d_in[4];
  const float* W2   = (const float*)d_in[5];
  const float* b2   = (const float*)d_in[6];
  const float* attn = (const float*)d_in[7];
  float* out = (float*)d_out;

  // Workspace: scores/exp-values [E], seg_max [N], denom [N]  (~2.4 MB total)
  float* ws       = (float*)d_ws;
  float* ws_score = ws;
  float* seg_max  = ws + N_EDGES;
  float* denom    = ws + N_EDGES + N_NODES;

  cfconv_init<<<(N_NODES * OUT_CH + 255) / 256, 256, 0, stream>>>(out, seg_max, denom);
  cfconv_heavy<0><<<2048, 256, 0, stream>>>(x, eidx, ea, W1, b1, W2, b2, attn,
                                            ws_score, seg_max, denom, out);
  cfconv_mid<<<(N_EDGES + 255) / 256, 256, 0, stream>>>(eidx, ws_score, seg_max, denom);
  cfconv_heavy<1><<<2048, 256, 0, stream>>>(x, eidx, ea, W1, b1, W2, b2, attn,
                                            ws_score, seg_max, denom, out);
}
